// SparseVLMModel_85693187490528
// MI455X (gfx1250) — matrix operand order repaired and checked by reading, not compiled, not run on hardware
//
#include <hip/hip_runtime.h>

// ---------------- problem constants (from reference) ----------------
#define B_     16
#define LV_    576
#define LT_    128
#define D_     1024
#define DLLM_  4096
#define K_KEEP_ 352      // LV - N_PRUNE
#define N_REC_  112
#define N_CLU_  56
#define K_TOT_  408      // K_KEEP + N_CLU
#define M_PAD_  416      // K_TOT padded to multiple of 16
#define OUT_L_  536      // K_TOT + LT
#define KT_     (D_ / 32)      // 32 k-tiles
#define NT_     (DLLM_ / 16)   // 256 n-tiles

// ---------------- WMMA fragment types ----------------
typedef __attribute__((ext_vector_type(16))) __bf16 v16bf;
typedef __attribute__((ext_vector_type(8)))  float  v8f;

union BF16x16 { v16bf v; unsigned short u[16]; uint4 q[2]; };

__device__ __forceinline__ unsigned short f2bf(float f) {
  __bf16 h = (__bf16)f;
  unsigned short u;
  __builtin_memcpy(&u, &h, 2);
  return u;
}

__device__ __forceinline__ float wave_sum(float v) {
#pragma unroll
  for (int m = 16; m >= 1; m >>= 1) v += __shfl_xor(v, m, 32);
  return v;
}
__device__ __forceinline__ float wave_max(float v) {
#pragma unroll
  for (int m = 16; m >= 1; m >>= 1) v = fmaxf(v, __shfl_xor(v, m, 32));
  return v;
}

// =====================================================================
// K0: pack W_proj (f32 row-major KxN) -> bf16 fragment-ordered tiles:
//     Wpack[kt][nt][lane][e], e contiguous 16 bf16 = 32B per lane.
// one wave per (kt,nt) tile; tiles = 32*256 = 8192 -> 1024 blocks x 256
// =====================================================================
__global__ void pack_w_kernel(const float* __restrict__ W,
                              unsigned short* __restrict__ Wp) {
  const int wave = threadIdx.x >> 5;
  const int tile = blockIdx.x * 8 + wave;       // kt*256 + nt
  const int kt = tile >> 8, nt = tile & 255;
  const int lane = threadIdx.x & 31, lo = lane & 15, hi = lane >> 4;
  const float* src = W + (size_t)(kt * 32 + hi * 16) * DLLM_ + nt * 16 + lo;
  unsigned short* dst = Wp + ((size_t)tile * 32 + lane) * 16;
#pragma unroll
  for (int e = 0; e < 16; ++e) dst[e] = f2bf(src[(size_t)e * DLLM_]);
}

// =====================================================================
// K1: attn[b,v,t] = sum_d patch[b,v,d]*text[b,t,d]   (bf16 WMMA, f32 acc)
// one wave per 16x16 tile; tiles = B * (LV/16) * (LT/16) = 16*36*8 = 4608
// =====================================================================
__global__ void attn_kernel(const float* __restrict__ patch,
                            const float* __restrict__ text,
                            float* __restrict__ attn) {
  const int wave = threadIdx.x >> 5;
  const int tile = blockIdx.x * 4 + wave;
  const int b  = tile / (36 * 8);
  const int r  = tile % (36 * 8);
  const int mt = r >> 3, nt = r & 7;
  const int lane = threadIdx.x & 31, lo = lane & 15, hi = lane >> 4;
  const int m0 = mt * 16, n0 = nt * 16;

  const float* arow = patch + ((size_t)b * LV_ + m0 + lo) * D_;
  const float* brow = text  + ((size_t)b * LT_ + n0 + lo) * D_;
  v8f c = {};
  for (int k0 = 0; k0 < D_; k0 += 32) {
    BF16x16 a, bb;
#pragma unroll
    for (int e = 0; e < 8; ++e) {
      a.u[e]     = f2bf(arow[k0 + hi * 16 + e]);
      a.u[e + 8] = f2bf(arow[k0 + hi * 16 + 8 + e]);
    }
#pragma unroll
    for (int e = 0; e < 16; ++e) bb.u[e] = f2bf(brow[k0 + hi * 16 + e]);
    c = __builtin_amdgcn_wmma_f32_16x16x32_bf16(false, a.v, false, bb.v,
                                                (short)0, c, false, false);
  }
  float* obase = attn + (size_t)b * LV_ * LT_;
#pragma unroll
  for (int rr = 0; rr < 8; ++rr) {
    int m = m0 + rr + 8 * hi;
    obase[(size_t)m * LT_ + n0 + lo] = c[rr];
  }
}

// =====================================================================
// K2: row softmax (in place), text_rel, raters, vision_scores
// one workgroup (256 thr = 8 waves) per batch
// =====================================================================
__global__ void softmax_rel_kernel(float* __restrict__ soft,
                                   float* __restrict__ scores) {
  const int b = blockIdx.x, tid = threadIdx.x;
  const int wave = tid >> 5, lane = tid & 31;
  __shared__ float colsum[LT_];
  __shared__ float red[LT_];
  __shared__ float maskw[LT_];
  __shared__ float s_invden;

  if (tid < LT_) colsum[tid] = 0.0f;
  __syncthreads();

  float* base = soft + (size_t)b * LV_ * LT_;
  for (int v = wave; v < LV_; v += 8) {
    float* row = base + (size_t)v * LT_;
    float x0 = row[lane], x1 = row[lane + 32], x2 = row[lane + 64], x3 = row[lane + 96];
    float m = fmaxf(fmaxf(x0, x1), fmaxf(x2, x3));
    m = wave_max(m);
    x0 = __expf(x0 - m); x1 = __expf(x1 - m);
    x2 = __expf(x2 - m); x3 = __expf(x3 - m);
    float s = wave_sum(x0 + x1 + x2 + x3);
    float inv = 1.0f / s;
    x0 *= inv; x1 *= inv; x2 *= inv; x3 *= inv;
    row[lane] = x0; row[lane + 32] = x1; row[lane + 64] = x2; row[lane + 96] = x3;
    atomicAdd(&colsum[lane], x0);       atomicAdd(&colsum[lane + 32], x1);
    atomicAdd(&colsum[lane + 64], x2);  atomicAdd(&colsum[lane + 96], x3);
  }
  __syncthreads();

  if (tid < LT_) red[tid] = colsum[tid];
  __syncthreads();
  for (int s = 64; s > 0; s >>= 1) { if (tid < s) red[tid] += red[tid + s]; __syncthreads(); }
  float thresh = red[0] / (float)LT_;
  if (tid < LT_) maskw[tid] = (colsum[tid] >= thresh) ? 1.0f : 0.0f;
  __syncthreads();
  if (tid < LT_) red[tid] = maskw[tid];
  __syncthreads();
  for (int s = 64; s > 0; s >>= 1) { if (tid < s) red[tid] += red[tid + s]; __syncthreads(); }
  if (tid == 0) s_invden = 1.0f / fmaxf(red[0], 1e-5f);
  __syncthreads();
  const float invden = s_invden;

  for (int v = wave; v < LV_; v += 8) {
    const float* row = base + (size_t)v * LT_;
    float s = row[lane] * maskw[lane] + row[lane + 32] * maskw[lane + 32]
            + row[lane + 64] * maskw[lane + 64] + row[lane + 96] * maskw[lane + 96];
    s = wave_sum(s);
    if (lane == 0) scores[b * LV_ + v] = s * invden;
  }
}

// =====================================================================
// K3: per-batch bitonic argsort (desc score, asc idx ties), gather:
//  - rows 0..351   -> topk (bf16)
//  - rows 352..463 -> normalized rec_n (f32)
//  - zero pad topk rows 408..415
// one workgroup (512 thr = 16 waves) per batch
// =====================================================================
__global__ void sort_prune_kernel(const float* __restrict__ scores,
                                  const float* __restrict__ patch,
                                  int* __restrict__ order,
                                  unsigned short* __restrict__ topk,
                                  float* __restrict__ recn) {
  const int b = blockIdx.x, tid = threadIdx.x;
  __shared__ float skey[1024];
  __shared__ int   sidx[1024];
  for (int i = tid; i < 1024; i += 512) {
    skey[i] = (i < LV_) ? scores[b * LV_ + i] : -3.4e38f;
    sidx[i] = i;
  }
  __syncthreads();
  for (int k = 2; k <= 1024; k <<= 1) {
    for (int j = k >> 1; j > 0; j >>= 1) {
      for (int i = tid; i < 1024; i += 512) {
        int ixj = i ^ j;
        if (ixj > i) {
          float ka = skey[i], kb = skey[ixj];
          int   ia = sidx[i], ib = sidx[ixj];
          bool before = (ka > kb) || (ka == kb && ia < ib);
          bool asc = ((i & k) == 0);
          if (asc != before) {
            skey[i] = kb; skey[ixj] = ka;
            sidx[i] = ib; sidx[ixj] = ia;
          }
        }
      }
      __syncthreads();
    }
  }
  for (int i = tid; i < LV_; i += 512) order[b * LV_ + i] = sidx[i];

  const int wave = tid >> 5, lane = tid & 31;    // 16 waves
  for (int r = wave; r < K_KEEP_; r += 16) {
    const float* src = patch + ((size_t)b * LV_ + sidx[r]) * D_;
    unsigned short* dst = topk + ((size_t)b * M_PAD_ + r) * D_;
    for (int c = lane; c < D_; c += 32) dst[c] = f2bf(src[c]);
  }
  for (int r = wave; r < N_REC_; r += 16) {
    const float* src = patch + ((size_t)b * LV_ + sidx[K_KEEP_ + r]) * D_;
    float ss = 0.f;
    for (int c = lane; c < D_; c += 32) { float x = src[c]; ss += x * x; }
    ss = wave_sum(ss);
    float inv = 1.0f / sqrtf(ss + 1e-12f);
    float* dst = recn + ((size_t)b * N_REC_ + r) * D_;
    for (int c = lane; c < D_; c += 32) dst[c] = src[c] * inv;
  }
  for (int i = tid; i < (M_PAD_ - K_TOT_) * D_; i += 512)
    topk[((size_t)b * M_PAD_ + K_TOT_) * D_ + i] = 0;
}

// =====================================================================
// K4: Gram = rec_n @ rec_nT  (112x112, K=1024), bf16 WMMA
// tiles = B * 7 * 7 = 784; one wave each
// =====================================================================
__global__ void gram_kernel(const float* __restrict__ recn,
                            float* __restrict__ gram) {
  const int wave = threadIdx.x >> 5;
  const int tile = blockIdx.x * 4 + wave;
  const int b = tile / 49, r = tile % 49;
  const int mt = r / 7, nt = r % 7;
  const int lane = threadIdx.x & 31, lo = lane & 15, hi = lane >> 4;
  const int m0 = mt * 16, n0 = nt * 16;

  const float* arow = recn + ((size_t)b * N_REC_ + m0 + lo) * D_;
  const float* brow = recn + ((size_t)b * N_REC_ + n0 + lo) * D_;
  v8f c = {};
  for (int k0 = 0; k0 < D_; k0 += 32) {
    BF16x16 a, bb;
#pragma unroll
    for (int e = 0; e < 8; ++e) {
      a.u[e]     = f2bf(arow[k0 + hi * 16 + e]);
      a.u[e + 8] = f2bf(arow[k0 + hi * 16 + 8 + e]);
    }
#pragma unroll
    for (int e = 0; e < 16; ++e) bb.u[e] = f2bf(brow[k0 + hi * 16 + e]);
    c = __builtin_amdgcn_wmma_f32_16x16x32_bf16(false, a.v, false, bb.v,
                                                (short)0, c, false, false);
  }
  float* g = gram + (size_t)b * N_REC_ * N_REC_;
#pragma unroll
  for (int rr = 0; rr < 8; ++rr)
    g[(size_t)(m0 + rr + 8 * hi) * N_REC_ + n0 + lo] = c[rr];
}

// =====================================================================
// K5: farthest-point clustering on Gram + segment-sum aggregation.
// one workgroup (128 thr) per batch; agg rows -> bf16 topk[352..407]
// =====================================================================
__global__ void cluster_kernel(const float* __restrict__ gram,
                               const int* __restrict__ order,
                               const float* __restrict__ patch,
                               unsigned short* __restrict__ topk) {
  const int b = blockIdx.x, t = threadIdx.x;
  __shared__ float mind[N_REC_];
  __shared__ int   cidx[N_CLU_];
  __shared__ int   assign_s[N_REC_];
  __shared__ float rv[128];
  __shared__ int   ri[128];
  __shared__ float aggs[N_CLU_ * 128];   // 28 KB

  const float* G = gram + (size_t)b * N_REC_ * N_REC_;
  if (t < N_REC_) mind[t] = 1.0f - G[(size_t)t * N_REC_ + 0];
  if (t == 0) cidx[0] = 0;
  __syncthreads();

  for (int j = 1; j < N_CLU_; ++j) {
    rv[t] = (t < N_REC_) ? mind[t] : -3.4e38f;
    ri[t] = t;
    __syncthreads();
    for (int s = 64; s > 0; s >>= 1) {
      if (t < s) {
        if (rv[t + s] > rv[t] || (rv[t + s] == rv[t] && ri[t + s] < ri[t])) {
          rv[t] = rv[t + s]; ri[t] = ri[t + s];
        }
      }
      __syncthreads();
    }
    int istar = ri[0];
    if (t == 0) cidx[j] = istar;
    if (t < N_REC_) mind[t] = fminf(mind[t], 1.0f - G[(size_t)t * N_REC_ + istar]);
    __syncthreads();
  }

  if (t < N_REC_) {
    float best = -3.4e38f; int bj = 0;
    for (int j = 0; j < N_CLU_; ++j) {
      float g = G[(size_t)t * N_REC_ + cidx[j]];
      if (g > best) { best = g; bj = j; }
    }
    assign_s[t] = bj;
  }
  __syncthreads();

  const int* ordv = order + b * LV_;
  for (int chunk = 0; chunk < D_ / 128; ++chunk) {
    const int d0 = chunk * 128;
    for (int i = t; i < N_CLU_ * 128; i += 128) aggs[i] = 0.0f;
    __syncthreads();
    for (int k = 0; k < N_REC_; ++k) {
      int pv = ordv[K_KEEP_ + k];
      float val = patch[((size_t)b * LV_ + pv) * D_ + d0 + t];
      aggs[assign_s[k] * 128 + t] += val;
    }
    __syncthreads();
    for (int j = 0; j < N_CLU_; ++j)
      topk[((size_t)b * M_PAD_ + K_KEEP_ + j) * D_ + d0 + t] = f2bf(aggs[j * 128 + t]);
    __syncthreads();
  }
}

// =====================================================================
// K6: visual = topk(bf16) @ W_proj + b_proj  -> d_out rows 0..407
// one wave per (1 M-tile x 4 N-tiles): A loaded once, 4 WMMAs/k-step.
// tiles = B * 26 * 64 = 26624 -> 3328 blocks x 256 (8 waves)
// =====================================================================
__global__ void proj_kernel(const unsigned short* __restrict__ topk,
                            const unsigned short* __restrict__ Wp,
                            const float* __restrict__ bias,
                            float* __restrict__ out) {
  const int wave = threadIdx.x >> 5;
  const int tile = blockIdx.x * 8 + wave;
  const int b   = tile / (26 * 64);
  const int rem = tile % (26 * 64);
  const int mt = rem >> 6, ng = rem & 63;       // ng: group of 4 n-tiles
  const int lane = threadIdx.x & 31, lo = lane & 15, hi = lane >> 4;
  const int m0 = mt * 16;

  const unsigned short* arow = topk + ((size_t)b * M_PAD_ + m0 + lo) * D_;
  v8f c0 = {}, c1 = {}, c2 = {}, c3 = {};
  for (int kt = 0; kt < KT_; ++kt) {
    const int k0 = kt * 32;
    BF16x16 a;
    a.q[0] = *(const uint4*)(arow + k0 + hi * 16);       // K = k0 + hi*16 + e
    a.q[1] = *(const uint4*)(arow + k0 + hi * 16 + 8);   // K = k0+hi*16+8+e
    const unsigned short* wp =
        Wp + (((size_t)kt * NT_ + ng * 4) * 32 + lane) * 16;
    if (kt + 1 < KT_)
      __builtin_prefetch(Wp + (((size_t)(kt + 1) * NT_ + ng * 4) * 32 + lane) * 16, 0, 3);
    BF16x16 b0, b1, b2, b3;
    b0.q[0] = *(const uint4*)(wp);           b0.q[1] = *(const uint4*)(wp + 8);
    b1.q[0] = *(const uint4*)(wp + 512);     b1.q[1] = *(const uint4*)(wp + 520);
    b2.q[0] = *(const uint4*)(wp + 1024);    b2.q[1] = *(const uint4*)(wp + 1032);
    b3.q[0] = *(const uint4*)(wp + 1536);    b3.q[1] = *(const uint4*)(wp + 1544);
    c0 = __builtin_amdgcn_wmma_f32_16x16x32_bf16(false, a.v, false, b0.v, (short)0, c0, false, false);
    c1 = __builtin_amdgcn_wmma_f32_16x16x32_bf16(false, a.v, false, b1.v, (short)0, c1, false, false);
    c2 = __builtin_amdgcn_wmma_f32_16x16x32_bf16(false, a.v, false, b2.v, (short)0, c2, false, false);
    c3 = __builtin_amdgcn_wmma_f32_16x16x32_bf16(false, a.v, false, b3.v, (short)0, c3, false, false);
  }
#pragma unroll
  for (int q = 0; q < 4; ++q) {
    const int n0 = (ng * 4 + q) * 16;
    const float bv = bias[n0 + lo];
    const v8f& c = (q == 0) ? c0 : (q == 1) ? c1 : (q == 2) ? c2 : c3;
#pragma unroll
    for (int rr = 0; rr < 8; ++rr) {
      int m = m0 + rr + 8 * hi;
      if (m < K_TOT_)
        out[((size_t)b * OUT_L_ + m) * DLLM_ + n0 + lo] = c[rr] + bv;
    }
  }
}

// =====================================================================
// K7: copy input_embeds (B,128,4096) into d_out rows 408..535
// =====================================================================
__global__ void copy_embeds_kernel(const float4* __restrict__ src,
                                   float* __restrict__ out) {
  const size_t total = (size_t)B_ * LT_ * DLLM_ / 4;
  for (size_t idx = (size_t)blockIdx.x * blockDim.x + threadIdx.x; idx < total;
       idx += (size_t)gridDim.x * blockDim.x) {
    size_t e = idx * 4;
    size_t b = e / ((size_t)LT_ * DLLM_);
    size_t rem = e % ((size_t)LT_ * DLLM_);
    float4 v = src[idx];
    *(float4*)(out + ((size_t)b * OUT_L_ + K_TOT_) * DLLM_ + rem) = v;
  }
}

// =====================================================================
extern "C" void kernel_launch(void* const* d_in, const int* in_sizes, int n_in,
                              void* d_out, int out_size, void* d_ws, size_t ws_size,
                              hipStream_t stream) {
  (void)in_sizes; (void)n_in; (void)out_size; (void)ws_size;
  const float* patch = (const float*)d_in[0];   // (B,LV,D)
  const float* text  = (const float*)d_in[1];   // (B,LT,D)
  const float* inemb = (const float*)d_in[2];   // (B,LT,DLLM)
  const float* Wproj = (const float*)d_in[3];   // (D,DLLM)
  const float* bproj = (const float*)d_in[4];   // (DLLM,)
  float* out = (float*)d_out;

  char* ws = (char*)d_ws;
  size_t off = 0;
  auto alignup = [](size_t x) { return (x + 255) & ~(size_t)255; };
  float* ws_soft = (float*)(ws + off);            off = alignup(off + (size_t)B_ * LV_ * LT_ * 4);
  float* ws_scores = (float*)(ws + off);          off = alignup(off + (size_t)B_ * LV_ * 4);
  int*   ws_order = (int*)(ws + off);             off = alignup(off + (size_t)B_ * LV_ * 4);
  float* ws_recn = (float*)(ws + off);            off = alignup(off + (size_t)B_ * N_REC_ * D_ * 4);
  float* ws_gram = (float*)(ws + off);            off = alignup(off + (size_t)B_ * N_REC_ * N_REC_ * 4);
  unsigned short* ws_topk = (unsigned short*)(ws + off);
  off = alignup(off + (size_t)B_ * M_PAD_ * D_ * 2);
  unsigned short* ws_wpack = (unsigned short*)(ws + off);
  off = alignup(off + (size_t)D_ * DLLM_ * 2);
  // total workspace ~35 MB

  pack_w_kernel<<<1024, 256, 0, stream>>>(Wproj, ws_wpack);
  attn_kernel<<<1152, 128, 0, stream>>>(patch, text, ws_soft);
  softmax_rel_kernel<<<B_, 256, 0, stream>>>(ws_soft, ws_scores);
  sort_prune_kernel<<<B_, 512, 0, stream>>>(ws_scores, patch, ws_order, ws_topk, ws_recn);
  gram_kernel<<<196, 128, 0, stream>>>(ws_recn, ws_gram);
  cluster_kernel<<<B_, 128, 0, stream>>>(ws_gram, ws_order, patch, ws_topk);
  proj_kernel<<<3328, 256, 0, stream>>>(ws_topk, ws_wpack, bproj, out);
  copy_embeds_kernel<<<4096, 256, 0, stream>>>((const float4*)inemb, out);
}
